// lstm_crf_74285754351619
// MI455X (gfx1250) — matrix operand order repaired and checked by reading, not compiled, not run on hardware
//
#include <hip/hip_runtime.h>
#include <stdint.h>

typedef __attribute__((ext_vector_type(16))) __bf16 v16bf;
typedef __attribute__((ext_vector_type(8)))  float  v8f;
typedef __attribute__((ext_vector_type(4)))  unsigned int u32x4;
typedef __attribute__((ext_vector_type(8)))  int i32x8;
typedef __attribute__((ext_vector_type(4)))  int i32x4;

static constexpr int B_    = 16;
static constexpr int T_    = 512;
static constexpr int E_    = 100;
static constexpr int HD    = 500;
static constexpr int HDP   = 512;    // padded hidden
static constexpr int NTG   = 128;    // ntiles over padded gate dim (4*512/16)
static constexpr int KT_H  = 16;     // ktiles for K=512
static constexpr int KT_E  = 4;      // ktiles for E padded to 128
static constexpr int KT_L  = 32;     // ktiles for concat K padded to 1024
static constexpr int NTAG_ = 32;
static constexpr float NEGV = -10000.0f;
static constexpr int SOS_  = 1;

__device__ __forceinline__ unsigned short f2bf(float f) {
  union { float f; uint32_t u; } v; v.f = f;
  uint32_t u = v.u;
  return (unsigned short)((u + 0x7FFFu + ((u >> 16) & 1u)) >> 16);
}

// Inverse of the CDNA5 16-bit A-matrix (16x32) lane/half layout:
// given (row m, k-offset ko in tile) -> (lane, half)
__device__ __forceinline__ void amap_inv(int m, int ko, int& lane, int& h) {
  if      (ko < 8)  { lane = m;      h = ko;      }
  else if (ko < 16) { lane = m + 16; h = ko - 8;  }
  else if (ko < 24) { lane = m;      h = ko - 8;  }
  else              { lane = m + 16; h = ko - 16; }
}

__device__ __forceinline__ v8f wmma_bf16(v16bf a, v16bf b, v8f c) {
  return __builtin_amdgcn_wmma_f32_16x16x32_bf16(false, a, false, b, (short)0, c,
                                                 false, false);
}

// ---------------------------------------------------------------------------
// CDNA5 async staging of a contiguous block (global -> LDS), shared by the
// whole workgroup. Preferred path: Tensor Data Mover with a 1-D descriptor
// (D# group0/group1 per ISA 8.3/8.4, data_size=8B, type=2), TENSORcnt wait.
// This toolchain exposes the 6-arg builtin (extra zero-filled i32x8 group).
// Fallback: GLOBAL_LOAD_ASYNC_TO_LDS_B128 + s_wait_asynccnt.
// ---------------------------------------------------------------------------
#if __has_builtin(__builtin_amdgcn_tensor_load_to_lds)
__device__ __forceinline__ void tdm_load_1d(uint32_t lds_addr, const void* gsrc,
                                            uint32_t n8 /* 8-byte units */) {
  uint64_t ga = (uint64_t)(uintptr_t)gsrc;
  u32x4 g0 = {0u, 0u, 0u, 0u};
  g0[0] = 1u;                                   // count=1, user descriptor
  g0[1] = lds_addr;                             // lds_addr (bytes)
  g0[2] = (uint32_t)ga;                         // global_addr[31:0]
  g0[3] = (uint32_t)((ga >> 32) & 0x01FFFFFFu)  // global_addr[56:32]
        | (2u << 30);                           // type=2 ("image")
  i32x8 g1 = {0, 0, 0, 0, 0, 0, 0, 0};
  g1[0] = (int)(3u << 16);                      // data_size=3 (8 bytes)
  g1[1] = (int)((n8 & 0xFFFFu) << 16);          // tensor_dim0[15:0]
  g1[2] = (int)(((n8 >> 16) & 0xFFFFu)          // tensor_dim0[31:16]
        | (1u << 16));                          // tensor_dim1 = 1
  g1[3] = (int)((n8 & 0xFFFFu) << 16);          // tile_dim0 = n8 (<=4096 here)
  g1[4] = 0;                                    // tile_dim1/2 unused
  g1[5] = (int)n8;                              // tensor_dim0_stride[31:0]
  g1[6] = 0;
  g1[7] = 0;
  i32x4 gz4 = {0, 0, 0, 0};
  i32x8 gz8 = {0, 0, 0, 0, 0, 0, 0, 0};
  __builtin_amdgcn_tensor_load_to_lds(g0, g1, gz4, gz4, gz8, 0);
}
#endif

__device__ __forceinline__ void stage_to_lds(unsigned short* sdst,
                                             const unsigned short* src,
                                             int bytes) {
#if __has_builtin(__builtin_amdgcn_tensor_load_to_lds)
  if (threadIdx.x == 0)
    tdm_load_1d((uint32_t)(uintptr_t)sdst, src, (uint32_t)(bytes >> 3));
#if __has_builtin(__builtin_amdgcn_s_wait_tensorcnt)
  __builtin_amdgcn_s_wait_tensorcnt(0);
#else
  asm volatile("s_wait_tensorcnt 0x0" ::: "memory");
#endif
#else
  uint32_t base = (uint32_t)(uintptr_t)sdst;
  for (int i = threadIdx.x * 16; i < bytes; i += blockDim.x * 16)
    asm volatile("global_load_async_to_lds_b128 %0, %1, off"
                 :: "v"(base + (uint32_t)i), "v"((const char*)src + i)
                 : "memory");
#if __has_builtin(__builtin_amdgcn_s_wait_asynccnt)
  __builtin_amdgcn_s_wait_asynccnt(0);
#else
  asm volatile("s_wait_asynccnt 0x0" ::: "memory");
#endif
#endif
  __syncthreads();
}

// ---------------------------------------------------------------------------
// Repack f32 weight matrix into bf16 WMMA B-fragment layout:
// dst[(nt*ktiles + kt)*512 + lane*16 + h]; element = W[rowmap(n)][colmap(k)]
// rowmode: 0 = direct (n<nrows), 1 = gate-padded (4 sections of 512 -> 500)
// colmode: 0 = direct (k<ncols), 1 = concat remap (1024 padded -> 2x500)
// ---------------------------------------------------------------------------
__global__ void repack_kernel(const float* __restrict__ W,
                              unsigned short* __restrict__ dst,
                              int ntiles, int ktiles, int rowmode, int nrows,
                              int colmode, int ncols) {
  int total = ntiles * ktiles * 512;
  for (int idx = blockIdx.x * blockDim.x + threadIdx.x; idx < total;
       idx += gridDim.x * blockDim.x) {
    int h    = idx & 15;
    int lane = (idx >> 4) & 31;
    int tile = idx >> 9;
    int kt   = tile % ktiles;
    int nt   = tile / ktiles;
    int n = nt * 16 + (lane & 15);
    int k = kt * 32 + ((h < 8) ? ((lane >> 4) * 8 + h) : (16 + (lane >> 4) * 8 + (h - 8)));
    int row;
    if (rowmode == 1) { int g = n >> 9, id = n & 511; row = (id < HD) ? g * HD + id : -1; }
    else              { row = (n < nrows) ? n : -1; }
    int col;
    if (colmode == 1) {
      if      (k < HD)        col = k;
      else if (k < HDP)       col = -1;
      else if (k < HDP + HD)  col = k - (HDP - HD);
      else                    col = -1;
    } else              { col = (k < ncols) ? k : -1; }
    float v = (row >= 0 && col >= 0) ? W[(size_t)row * ncols + col] : 0.0f;
    dst[idx] = f2bf(v);
  }
}

// ---------------------------------------------------------------------------
// Embedding gather -> bf16 A-fragments  xsfrag[t][kt(4)][lane][half]
// ---------------------------------------------------------------------------
__global__ void embed_kernel(const int* __restrict__ x,
                             const float* __restrict__ embed_W,
                             unsigned short* __restrict__ xsfrag) {
  int t = blockIdx.x;
  for (int e = threadIdx.x; e < KT_E * 512; e += blockDim.x) {
    int kt = e >> 9;
    int r  = e & 511;
    int m  = r >> 5;
    int ko = r & 31;
    int k  = kt * 32 + ko;
    float v = 0.f;
    if (k < E_) {
      int tok = x[m * T_ + t];
      v = embed_W[(size_t)tok * E_ + k];
    }
    int lane, h; amap_inv(m, ko, lane, h);
    xsfrag[(((size_t)t * KT_E + kt) * 32 + lane) * 16 + h] = f2bf(v);
  }
}

// ---------------------------------------------------------------------------
// Gin = A(t) @ Bfrag + (bias1+bias2). A(t) is shared by all 8 waves, so it is
// staged once into LDS via TDM/async copy; B streams from L2 as b128 loads.
// Output in (t, nt, lane, 8) f32 fragment layout for direct v8f accumulator
// reload in the recurrent kernel. grid = (T, 16), block = 256 (8 waves).
// ---------------------------------------------------------------------------
__global__ void __launch_bounds__(256)
gin_gemm_kernel(const unsigned short* __restrict__ afrag, int ktiles,
                const unsigned short* __restrict__ bfrag,
                const float* __restrict__ bias1, const float* __restrict__ bias2,
                float* __restrict__ ginfrag) {
  __shared__ __align__(128) unsigned short sAf[KT_L * 512];  // up to 32KB
  int t    = blockIdx.x;
  int wave = threadIdx.x >> 5;
  int lane = threadIdx.x & 31;
  int nt   = blockIdx.y * 8 + wave;  // 0..127

  stage_to_lds(sAf, afrag + (size_t)t * ktiles * 512, ktiles * 1024);

  v8f acc = {};
  const v16bf* ap = (const v16bf*)sAf;
  const v16bf* bp = (const v16bf*)(bfrag + (size_t)nt * ktiles * 512);
  for (int kt = 0; kt < ktiles; ++kt) {
    v16bf a = ap[kt * 32 + lane];
    v16bf b = bp[kt * 32 + lane];
    acc = wmma_bf16(a, b, acc);
  }
  int n = nt * 16 + (lane & 15);
  int g = n >> 9, id = n & 511;
  float bv = (id < HD) ? (bias1[g * HD + id] + bias2[g * HD + id]) : 0.f;
#pragma unroll
  for (int e = 0; e < 8; ++e) acc[e] += bv;
  *(v8f*)(ginfrag + (((size_t)t * NTG + nt) * 32 + lane) * 8) = acc;
}

// ---------------------------------------------------------------------------
// Recurrent LSTM (one layer, both directions; blockIdx.x = dir).
// 32 waves; wave w owns hidden units [16w,16w+16) and computes its i/f/g/o
// gate tiles itself (gate-aligned padded layout -> no cross-wave exchange).
// h kept in LDS pre-swizzled as WMMA A-fragments; c in LDS f32.
// ---------------------------------------------------------------------------
__global__ void __launch_bounds__(1024)
lstm_rec_kernel(const float* __restrict__ gin,          // (2,T,128,32,8) f32
                const unsigned short* __restrict__ whh, // (2,128,16,32,16) bf16
                const float* __restrict__ h0,           // (2,B,HD) this layer
                const float* __restrict__ c0,
                const int* __restrict__ x,              // (B,T)
                unsigned short* __restrict__ outfrag)   // (T,32,32,16) bf16
{
  const int dir  = blockIdx.x;
  const int wave = threadIdx.x >> 5;
  const int lane = threadIdx.x & 31;
  const bool rev = (dir == 1);
  const float* ginD = gin + (size_t)dir * T_ * NTG * 32 * 8;
  const unsigned short* whhD = whh + (size_t)dir * NTG * KT_H * 512;
  const float* h0D = h0 + dir * B_ * HD;
  const float* c0D = c0 + dir * B_ * HD;
  const int koffset = dir * HDP;  // fwd features at k=j, bwd at k=512+j

  __shared__ __align__(32) unsigned short sA[KT_H * 32 * 16]; // 16KB h fragments
  __shared__ float sC[B_ * HDP];                              // 32KB cell state

  for (int idx = threadIdx.x; idx < B_ * HDP; idx += blockDim.x) {
    int m = idx >> 9;
    int j = idx & 511;
    float hv = (j < HD) ? h0D[m * HD + j] : 0.f;
    float cv = (j < HD) ? c0D[m * HD + j] : 0.f;
    sC[m * HDP + j] = cv;
    int kt = j >> 5, ko = j & 31, l2, hh;
    amap_inv(m, ko, l2, hh);
    sA[(kt * 32 + l2) * 16 + hh] = f2bf(hv);
  }
  __syncthreads();

  const int jbase = wave * 16 + (lane & 15);  // hidden unit owned by this column

  for (int step = 0; step < T_; ++step) {
    int t = rev ? (T_ - 1 - step) : step;

    // accumulators start from precomputed input projection + biases
    v8f acc[4];
#pragma unroll
    for (int g = 0; g < 4; ++g)
      acc[g] = *(const v8f*)(ginD + (((size_t)t * NTG + g * 32 + wave) * 32 + lane) * 8);

    const v16bf* aP = (const v16bf*)sA;
    for (int kt = 0; kt < KT_H; ++kt) {
      v16bf a = aP[kt * 32 + lane];
      if (kt + 1 < KT_H)  // hint next weight slice (L2-resident) into L0
        __builtin_prefetch(whhD + ((size_t)wave * KT_H + kt + 1) * 512, 0, 3);
#pragma unroll
      for (int g = 0; g < 4; ++g) {
        const v16bf* bP =
            (const v16bf*)(whhD + ((size_t)(g * 32 + wave) * KT_H + kt) * 512);
        v16bf b = bP[lane];
        acc[g] = wmma_bf16(a, b, acc[g]);
      }
    }

    // pointwise cell update: element e -> batch m, hidden unit jbase
    float hnew[8]; bool msk[8]; float cnew[8];
#pragma unroll
    for (int e = 0; e < 8; ++e) {
      int m = e + ((lane >> 4) << 3);
      bool mk = (jbase < HD) && (x[m * T_ + t] > 0);
      float cprev = sC[m * HDP + jbase];
      float gi = acc[0][e], gf = acc[1][e], gg = acc[2][e], go = acc[3][e];
      float si = 1.f / (1.f + __expf(-gi));
      float sf = 1.f / (1.f + __expf(-gf));
      float so = 1.f / (1.f + __expf(-go));
      float tg = tanhf(gg);
      float cn = sf * cprev + si * tg;
      float hn = so * tanhf(cn);
      msk[e]  = mk;
      cnew[e] = mk ? cn : cprev;
      hnew[e] = hn;
    }

    __syncthreads();  // all waves done reading sA for this step

#pragma unroll
    for (int e = 0; e < 8; ++e) {
      int m = e + ((lane >> 4) << 3);
      int kt = jbase >> 5, ko = jbase & 31, l2, hh;
      amap_inv(m, ko, l2, hh);
      int sidx = (kt * 32 + l2) * 16 + hh;
      unsigned short oldb = sA[sidx];
      sA[sidx] = msk[e] ? f2bf(hnew[e]) : oldb;
      sC[m * HDP + jbase] = cnew[e];
      // masked output, written directly in A-fragment layout for next GEMMs
      float yv = msk[e] ? hnew[e] : 0.f;
      int k2 = koffset + jbase;
      int kt2 = k2 >> 5, ko2 = k2 & 31, l2b, h2b;
      amap_inv(m, ko2, l2b, h2b);
      outfrag[(((size_t)t * KT_L + kt2) * 32 + l2b) * 16 + h2b] = f2bf(yv);
    }
    __syncthreads();
  }
}

// ---------------------------------------------------------------------------
// Projection: y[t][b][tag] = mask * (l2(t) @ Wout.T + b_out). grid=T, 2 waves.
// A(t) staged into LDS once per block (shared by both waves).
// ---------------------------------------------------------------------------
__global__ void __launch_bounds__(64)
proj_kernel(const unsigned short* __restrict__ l2frag,
            const unsigned short* __restrict__ woutfrag,
            const float* __restrict__ b_out, const int* __restrict__ x,
            float* __restrict__ y) {
  __shared__ __align__(128) unsigned short sAf[KT_L * 512];  // 32KB
  int t    = blockIdx.x;
  int wave = threadIdx.x >> 5;
  int lane = threadIdx.x & 31;

  stage_to_lds(sAf, l2frag + (size_t)t * KT_L * 512, KT_L * 1024);

  v8f acc = {};
  const v16bf* ap = (const v16bf*)sAf;
  const v16bf* bp = (const v16bf*)(woutfrag + (size_t)wave * KT_L * 512);
  for (int kt = 0; kt < KT_L; ++kt)
    acc = wmma_bf16(ap[kt * 32 + lane], bp[kt * 32 + lane], acc);
  int n = wave * 16 + (lane & 15);
  float bv = b_out[n];
#pragma unroll
  for (int e = 0; e < 8; ++e) {
    int m = e + ((lane >> 4) << 3);
    bool mk = x[m * T_ + t] > 0;
    y[((size_t)t * B_ + m) * NTAG_ + n] = mk ? (acc[e] + bv) : 0.f;
  }
}

// ---------------------------------------------------------------------------
// CRF: gold score + forward algorithm. 16 waves (one per batch), lane = tag.
// ---------------------------------------------------------------------------
__global__ void __launch_bounds__(512)
crf_kernel(const float* __restrict__ y, const float* __restrict__ trans,
           const int* __restrict__ x, const int* __restrict__ y0,
           float* __restrict__ out) {
  __shared__ float sT[NTAG_ * NTAG_];
  __shared__ float sRes[B_];
  for (int i = threadIdx.x; i < NTAG_ * NTAG_; i += blockDim.x) sT[i] = trans[i];
  __syncthreads();
  int b    = threadIdx.x >> 5;
  int lane = threadIdx.x & 31;

  // gold = sum_t emit(y0[t]) + mask*trans[y0[t], prev]
  float acc = 0.f;
  for (int t = lane; t < T_; t += 32) {
    int tag  = y0[b * T_ + t];
    float em = y[((size_t)t * B_ + b) * NTAG_ + tag];
    int prev = (t == 0) ? SOS_ : y0[b * T_ + t - 1];
    float mk = (x[b * T_ + t] > 0) ? 1.f : 0.f;
    acc += em + mk * sT[tag * NTAG_ + prev];
  }
#pragma unroll
  for (int off = 16; off >= 1; off >>= 1) acc += __shfl_xor(acc, off, 32);
  float gold = acc;

  // forward scan
  float score = (lane == SOS_) ? 0.f : NEGV;
  for (int t = 0; t < T_; ++t) {
    float emit = y[((size_t)t * B_ + b) * NTAG_ + lane];
    float vmax = -1e30f;
    for (int j = 0; j < NTAG_; ++j) {
      float sj = __shfl(score, j, 32);
      vmax = fmaxf(vmax, sj + sT[lane * NTAG_ + j]);
    }
    float s = 0.f;
    for (int j = 0; j < NTAG_; ++j) {
      float sj = __shfl(score, j, 32);
      s += __expf(sj + sT[lane * NTAG_ + j] - vmax);
    }
    float newv = emit + vmax + __logf(s);
    float smax = score;
#pragma unroll
    for (int off = 16; off >= 1; off >>= 1)
      smax = fmaxf(smax, __shfl_xor(smax, off, 32));
    bool mk = x[b * T_ + t] > 0;
    score = mk ? newv : smax;
  }
  // Z = logsumexp over tags
  float mx = score;
#pragma unroll
  for (int off = 16; off >= 1; off >>= 1) mx = fmaxf(mx, __shfl_xor(mx, off, 32));
  float s = __expf(score - mx);
#pragma unroll
  for (int off = 16; off >= 1; off >>= 1) s += __shfl_xor(s, off, 32);
  float Z = mx + __logf(s);
  if (lane == 0) sRes[b] = Z - gold;
  __syncthreads();
  if (threadIdx.x == 0) {
    float m = 0.f;
    for (int i = 0; i < B_; ++i) m += sRes[i];
    out[0] = m / (float)B_;
  }
}

// ---------------------------------------------------------------------------
extern "C" void kernel_launch(void* const* d_in, const int* in_sizes, int n_in,
                              void* d_out, int out_size, void* d_ws, size_t ws_size,
                              hipStream_t stream) {
  const int*   x       = (const int*)d_in[0];
  const int*   y0      = (const int*)d_in[1];
  const float* embed_W = (const float*)d_in[2];
  const float* Wih0    = (const float*)d_in[3];
  const float* Whh0    = (const float*)d_in[4];
  const float* bih0    = (const float*)d_in[5];
  const float* bhh0    = (const float*)d_in[6];
  const float* Wih1    = (const float*)d_in[7];
  const float* Whh1    = (const float*)d_in[8];
  const float* bih1    = (const float*)d_in[9];
  const float* bhh1    = (const float*)d_in[10];
  const float* W_out   = (const float*)d_in[11];
  const float* b_out   = (const float*)d_in[12];
  const float* trans   = (const float*)d_in[13];
  const float* h0      = (const float*)d_in[14];
  const float* c0      = (const float*)d_in[15];

  // workspace carve-up (256B aligned)
  char* ws = (char*)d_ws;
  size_t off = 0;
  auto alloc = [&](size_t bytes) {
    size_t o = off;
    off += (bytes + 255) & ~(size_t)255;
    return o;
  };
  unsigned short* xsfrag   = (unsigned short*)(ws + alloc((size_t)T_ * KT_E * 512 * 2));
  unsigned short* l1frag   = (unsigned short*)(ws + alloc((size_t)T_ * KT_L * 512 * 2));
  unsigned short* l2frag   = (unsigned short*)(ws + alloc((size_t)T_ * KT_L * 512 * 2));
  float*          gin0     = (float*)(ws + alloc((size_t)2 * T_ * NTG * 256 * 4));
  float*          gin1     = (float*)(ws + alloc((size_t)2 * T_ * NTG * 256 * 4));
  unsigned short* whh0frag = (unsigned short*)(ws + alloc((size_t)2 * NTG * KT_H * 512 * 2));
  unsigned short* whh1frag = (unsigned short*)(ws + alloc((size_t)2 * NTG * KT_H * 512 * 2));
  unsigned short* wih0frag = (unsigned short*)(ws + alloc((size_t)2 * NTG * KT_E * 512 * 2));
  unsigned short* wih1frag = (unsigned short*)(ws + alloc((size_t)2 * NTG * KT_L * 512 * 2));
  unsigned short* woutfrag = (unsigned short*)(ws + alloc((size_t)2 * KT_L * 512 * 2));
  float*          ybuf     = (float*)(ws + alloc((size_t)T_ * B_ * NTAG_ * 4));
  (void)ws_size; (void)in_sizes; (void)n_in; (void)out_size;

  auto rpgrid = [](int ntiles, int ktiles) {
    return dim3((unsigned)((ntiles * ktiles * 512 + 255) / 256));
  };

  // --- one-time weight repacks into WMMA B-fragment layout (bf16) ---
  for (int d = 0; d < 2; ++d) {
    repack_kernel<<<rpgrid(NTG, KT_H), 256, 0, stream>>>(
        Whh0 + (size_t)d * 2000 * 500, whh0frag + (size_t)d * NTG * KT_H * 512,
        NTG, KT_H, 1, 2000, 0, 500);
    repack_kernel<<<rpgrid(NTG, KT_H), 256, 0, stream>>>(
        Whh1 + (size_t)d * 2000 * 500, whh1frag + (size_t)d * NTG * KT_H * 512,
        NTG, KT_H, 1, 2000, 0, 500);
    repack_kernel<<<rpgrid(NTG, KT_E), 256, 0, stream>>>(
        Wih0 + (size_t)d * 2000 * 100, wih0frag + (size_t)d * NTG * KT_E * 512,
        NTG, KT_E, 1, 2000, 0, 100);
    repack_kernel<<<rpgrid(NTG, KT_L), 256, 0, stream>>>(
        Wih1 + (size_t)d * 2000 * 1000, wih1frag + (size_t)d * NTG * KT_L * 512,
        NTG, KT_L, 1, 2000, 1, 1000);
  }
  repack_kernel<<<rpgrid(2, KT_L), 256, 0, stream>>>(
      W_out, woutfrag, 2, KT_L, 0, NTAG_, 1, 1000);

  // --- embedding gather into A-fragments ---
  embed_kernel<<<T_, 256, 0, stream>>>(x, embed_W, xsfrag);

  // --- layer 0: input projections (parallel GEMM), then recurrence ---
  for (int d = 0; d < 2; ++d)
    gin_gemm_kernel<<<dim3(T_, 16), 256, 0, stream>>>(
        xsfrag, KT_E, wih0frag + (size_t)d * NTG * KT_E * 512,
        bih0 + (size_t)d * 2000, bhh0 + (size_t)d * 2000,
        gin0 + (size_t)d * T_ * NTG * 256);
  lstm_rec_kernel<<<2, 1024, 0, stream>>>(gin0, whh0frag, h0, c0, x, l1frag);

  // --- layer 1 ---
  for (int d = 0; d < 2; ++d)
    gin_gemm_kernel<<<dim3(T_, 16), 256, 0, stream>>>(
        l1frag, KT_L, wih1frag + (size_t)d * NTG * KT_L * 512,
        bih1 + (size_t)d * 2000, bhh1 + (size_t)d * 2000,
        gin1 + (size_t)d * T_ * NTG * 256);
  lstm_rec_kernel<<<2, 1024, 0, stream>>>(gin1, whh1frag, h0 + 2 * B_ * HD,
                                          c0 + 2 * B_ * HD, x, l2frag);

  // --- projection + CRF ---
  proj_kernel<<<T_, 64, 0, stream>>>(l2frag, woutfrag, b_out, x, ybuf);
  crf_kernel<<<1, 512, 0, stream>>>(ybuf, trans, x, y0, (float*)d_out);
}
